// MinGRUSynthetic_37245956391285
// MI455X (gfx1250) — matrix operand order repaired and checked by reading, not compiled, not run on hardware
//
#include <hip/hip_runtime.h>
#include <hip/hip_bf16.h>

typedef __attribute__((ext_vector_type(16))) __bf16 v16bf;
typedef __attribute__((ext_vector_type(8)))  __bf16 v8bf;
typedef __attribute__((ext_vector_type(8)))  float  v8f;

namespace {
constexpr int B = 8, S = 4096, D = 512, L = 4, C = 8;
constexpr int NT = D / 16;   // 32 n-tiles per row of weights
constexpr int KT = D / 32;   // 16 K-steps of 32
constexpr int NC = 32;       // scan chunks
constexpr int CH = S / NC;   // 128 steps per chunk
}

union V16U { v16bf v; v8bf h[2]; };
union V8FU { v8f v; float f[8]; };
union PairU { unsigned u; __bf16 h[2]; };   // packed (a_t, b_t) gate pair

// ---- 1. embedding gather + bf16 convert ------------------------------------
__global__ __launch_bounds__(256) void embed_kernel(const int* __restrict__ x,
    const float* __restrict__ emb, __bf16* __restrict__ hbuf) {
  int t = blockIdx.x * 256 + threadIdx.x;      // over B*S*D
  int d = t & (D - 1);
  int bs = t / D;
  int tok = x[bs];
  hbuf[t] = (__bf16)emb[(size_t)tok * D + d];
}

// ---- 2. pack Wz/Wh into per-lane WMMA B-fragment order ---------------------
// dest layout: [l][m(z/h)][ntile][kt][lane][e]  (identity with flat t)
// B-fragment element mapping: k = kt*32 + (lane/16)*16 + e, n = ntile*16 + lane%16
__global__ __launch_bounds__(256) void pack_kernel(const float* __restrict__ Wz,
    const float* __restrict__ Wh, __bf16* __restrict__ wpack) {
  int t = blockIdx.x * 256 + threadIdx.x;      // over L*2*D*D
  int e    = t & 15;
  int lane = (t >> 4) & 31;
  int kt   = (t >> 9) & (KT - 1);
  int nt   = (t >> 13) & (NT - 1);
  int m    = (t >> 18) & 1;
  int l    = t >> 19;
  int k = kt * 32 + (lane >> 4) * 16 + e;
  int n = nt * 16 + (lane & 15);
  const float* W = (m == 0) ? Wz : Wh;
  wpack[t] = (__bf16)W[((size_t)l * D + k) * D + n];
}

// ---- 3. fused dual GEMM (z-gate + h-tilde) via WMMA bf16 -------------------
// wave computes 64 rows x 16 cols: 4 M-tiles, K-loop of 16 x (2 wmma per tile)
// epilogue packs (a, b) = (1-z, z*h~) into one 32-bit bf16 pair per element
__global__ __launch_bounds__(256) void gemm_gate_kernel(
    const __bf16* __restrict__ hbuf, const __bf16* __restrict__ wl,
    const float* __restrict__ bz, const float* __restrict__ bh,
    unsigned* __restrict__ abbuf) {
  int wave = blockIdx.x * 8 + (threadIdx.x >> 5);
  int lane = threadIdx.x & 31;
  int ntile  = wave & (NT - 1);
  int msuper = wave >> 5;                       // 512 supertiles of 64 rows
  int col = ntile * 16 + (lane & 15);
  int rowbase = msuper * 64 + (lane & 15);
  int khA = (lane >> 4) * 16;                   // A-layout K sub-offset per lane half

  const __bf16* wz = wl + (size_t)ntile * (KT * 512) + lane * 16;
  const __bf16* wh = wz + (size_t)NT * (KT * 512);

  v8f accZ[4] = {v8f{}, v8f{}, v8f{}, v8f{}};
  v8f accH[4] = {v8f{}, v8f{}, v8f{}, v8f{}};

  for (int kt = 0; kt < KT; ++kt) {
    v16bf bfz = *(const v16bf*)(wz + (size_t)kt * 512);
    v16bf bfh = *(const v16bf*)(wh + (size_t)kt * 512);
    int kbase = kt * 32 + khA;
#pragma unroll
    for (int mt = 0; mt < 4; ++mt) {
      // A in contiguous k order, as packed B: elems 0-15 -> K = kbase+0..15
      const __bf16* ap = hbuf + (size_t)(rowbase + mt * 16) * D + kbase;
      V16U a;
      a.h[0] = *(const v8bf*)ap;
      a.h[1] = *(const v8bf*)(ap + 8);
      accZ[mt] = __builtin_amdgcn_wmma_f32_16x16x32_bf16(
          false, a.v, false, bfz, (short)0, accZ[mt], false, false);
      accH[mt] = __builtin_amdgcn_wmma_f32_16x16x32_bf16(
          false, a.v, false, bfh, (short)0, accH[mt], false, false);
    }
  }

  float bzc = bz[col];
  float bhc = bh[col];
#pragma unroll
  for (int mt = 0; mt < 4; ++mt) {
    V8FU cz, ch; cz.v = accZ[mt]; ch.v = accH[mt];
    int row0 = msuper * 64 + mt * 16 + (lane >> 4) * 8;
#pragma unroll
    for (int r = 0; r < 8; ++r) {
      float zl = cz.f[r] + bzc;
      float z  = 1.0f / (1.0f + __expf(-zl));
      float ht = ch.f[r] + bhc;
      PairU p;
      p.h[0] = (__bf16)(1.0f - z);   // a_t
      p.h[1] = (__bf16)(z * ht);     // b_t = z_t * h~_t
      abbuf[(size_t)(row0 + r) * D + col] = p.u;
    }
  }
}

// ---- 4a. per-chunk affine summaries (h' = A*h + Bv) ------------------------
__global__ __launch_bounds__(256) void scan_sum_kernel(
    const unsigned* __restrict__ abbuf,
    float* __restrict__ Asum, float* __restrict__ Bsum) {
  int t = blockIdx.x * 256 + threadIdx.x;      // over B*NC*D, t=(b*NC+c)*D+d
  int d = t & (D - 1);
  int c = (t / D) & (NC - 1);
  int b = t / (D * NC);
  size_t base = ((size_t)b * S + (size_t)c * CH) * D + d;
  float A = 1.0f, Bv = 0.0f;
  for (int i = 0; i < CH; ++i) {
    PairU p; p.u = abbuf[base + (size_t)i * D];
    float av = (float)p.h[0];
    float bv = (float)p.h[1];
    A = av * A;
    Bv = av * Bv + bv;
  }
  Asum[t] = A;
  Bsum[t] = Bv;
}

// ---- 4b. sequential prefix over chunk summaries ----------------------------
__global__ __launch_bounds__(256) void scan_prefix_kernel(
    const float* __restrict__ Asum, const float* __restrict__ Bsum,
    float* __restrict__ hstart) {
  int t = blockIdx.x * 256 + threadIdx.x;      // over B*D
  int d = t & (D - 1);
  int b = t / D;
  float h = 0.0f;                              // h_0 = 0
  for (int c = 0; c < NC; ++c) {
    size_t i = ((size_t)b * NC + c) * D + d;
    hstart[i] = h;
    h = Asum[i] * h + Bsum[i];
  }
}

// ---- 4c. apply scan within chunks, emit bf16 activations -------------------
__global__ __launch_bounds__(256) void scan_apply_kernel(
    const unsigned* __restrict__ abbuf,
    const float* __restrict__ hstart, __bf16* __restrict__ hbuf) {
  int t = blockIdx.x * 256 + threadIdx.x;      // over B*NC*D
  int d = t & (D - 1);
  int c = (t / D) & (NC - 1);
  int b = t / (D * NC);
  size_t base = ((size_t)b * S + (size_t)c * CH) * D + d;
  float h = hstart[t];
  for (int i = 0; i < CH; ++i) {
    size_t o = base + (size_t)i * D;
    PairU p; p.u = abbuf[o];
    h = (float)p.h[0] * h + (float)p.h[1];
    hbuf[o] = (__bf16)h;
  }
}

// ---- 5. classifier on last timestep ----------------------------------------
__global__ void classify_kernel(const __bf16* __restrict__ hbuf,
    const float* __restrict__ Wo, const float* __restrict__ bo,
    float* __restrict__ out) {
  int t = threadIdx.x;                         // 64 threads, t = b*C + c
  int b = t / C;
  int c = t & (C - 1);
  const __bf16* hl = hbuf + ((size_t)b * S + (S - 1)) * D;
  float acc = bo[c];
  for (int d = 0; d < D; ++d)
    acc += (float)hl[d] * Wo[(size_t)d * C + c];
  out[t] = acc;
}

extern "C" void kernel_launch(void* const* d_in, const int* in_sizes, int n_in,
                              void* d_out, int out_size, void* d_ws, size_t ws_size,
                              hipStream_t stream) {
  const int*   x   = (const int*)d_in[0];
  const float* emb = (const float*)d_in[1];
  const float* Wz  = (const float*)d_in[2];
  const float* bz  = (const float*)d_in[3];
  const float* Wh  = (const float*)d_in[4];
  const float* bh  = (const float*)d_in[5];
  const float* Wo  = (const float*)d_in[6];
  const float* bo  = (const float*)d_in[7];
  float* out = (float*)d_out;

  char* ws = (char*)d_ws;
  size_t off = 0;
  __bf16*   hbuf  = (__bf16*)  (ws + off); off += (size_t)B * S * D * sizeof(__bf16);   // 33.5 MB
  unsigned* abbuf = (unsigned*)(ws + off); off += (size_t)B * S * D * sizeof(unsigned); // 67 MB
  __bf16*   wpack = (__bf16*)  (ws + off); off += (size_t)L * 2 * D * D * sizeof(__bf16); // 4 MB
  float*    Asum  = (float*)   (ws + off); off += (size_t)B * NC * D * sizeof(float);
  float*    Bsum  = (float*)   (ws + off); off += (size_t)B * NC * D * sizeof(float);
  float*    hst   = (float*)   (ws + off); off += (size_t)B * NC * D * sizeof(float);

  embed_kernel<<<(B * S * D) / 256, 256, 0, stream>>>(x, emb, hbuf);
  pack_kernel<<<(L * 2 * D * D) / 256, 256, 0, stream>>>(Wz, Wh, wpack);

  for (int l = 0; l < L; ++l) {
    const __bf16* wl = wpack + (size_t)l * 2 * D * D;
    gemm_gate_kernel<<<((B * S / 64) * NT) / 8, 256, 0, stream>>>(
        hbuf, wl, bz + (size_t)l * D, bh + (size_t)l * D, abbuf);
    scan_sum_kernel<<<(B * NC * D) / 256, 256, 0, stream>>>(abbuf, Asum, Bsum);
    scan_prefix_kernel<<<(B * D) / 256, 256, 0, stream>>>(Asum, Bsum, hst);
    scan_apply_kernel<<<(B * NC * D) / 256, 256, 0, stream>>>(abbuf, hst, hbuf);
  }

  classify_kernel<<<1, B * C, 0, stream>>>(hbuf, Wo, bo, out);
}
